// GraphAttentionHead_57947698758294
// MI455X (gfx1250) — matrix operand order repaired and checked by reading, not compiled, not run on hardware
//
#include <hip/hip_runtime.h>

#define LRELU_ALPHA 0.2f

typedef __attribute__((ext_vector_type(16))) __bf16 v16bf;
typedef __attribute__((ext_vector_type(8)))  __bf16 v8bf;
typedef __attribute__((ext_vector_type(8)))  float  v8f;
typedef __attribute__((ext_vector_type(8)))  float  v8fw;

// monotone float <-> uint mapping for atomicMax-based segment max
__device__ __forceinline__ unsigned f2ord(float f) {
  unsigned u = __float_as_uint(f);
  return (u & 0x80000000u) ? ~u : (u | 0x80000000u);
}
__device__ __forceinline__ float ord2f(unsigned t) {
  return (t & 0x80000000u) ? __uint_as_float(t & 0x7FFFFFFFu) : __uint_as_float(~t);
}

// ---------------- Kernel 1: Wh = h@W + b  (bf16 WMMA), fused f1/f2 ---------
// Block = 256 threads = 8 waves; wave w computes rows [blk*128 + w*16, +16) x 64 cols.
__global__ __launch_bounds__(256) void gat_gemm_wmma(
    const float* __restrict__ h, const float* __restrict__ W,
    const float* __restrict__ b, const float* __restrict__ a_src,
    const float* __restrict__ a_dest,
    float* __restrict__ Wh, float* __restrict__ f1, float* __restrict__ f2,
    int N)
{
  // W staged once per block in WMMA B-fragment layout:
// fragment (kchunk c, tile t), lane l holds 16 bf16: K = c*32 + (j/8)*16 + (l/16)*8 + (j&7),
// col = t*16 + (l&15), j = 0..15.
  __shared__ __align__(32) __bf16 sB[8 * 4 * 32 * 16]; // 32 KB
  const int tid = threadIdx.x;
  for (int idx = tid; idx < 8 * 4 * 32 * 16; idx += 256) {
    int j = idx & 15;
    int l = (idx >> 4) & 31;
    int t = (idx >> 9) & 3;
    int c = idx >> 11;
    int k   = c * 32 + ((j >> 3) << 4) + ((l >> 4) << 3) + (j & 7);
    int col = t * 16 + (l & 15);
    sB[idx] = (__bf16)W[k * 64 + col];   // hardware v_cvt, RNE
  }
  __syncthreads();

  const int wave = tid >> 5;
  const int lane = tid & 31;
  const int m    = lane & 15;   // row (A) / col (B,C) within tile
  const int hi   = lane >> 4;   // half-wave select
  const int row0 = blockIdx.x * 128 + wave * 16;
  const int row  = row0 + m;
  const int rowc = row < N ? row : (N - 1); // clamp loads; stores masked
  const float* hrow = h + (long long)rowc * 256;

  const v16bf* bp = reinterpret_cast<const v16bf*>(sB);
  v8f acc[4] = {};

  const int base = hi * 8; // A-layout: lanes 0-15 -> K base 0, lanes 16-31 -> base 8
  const v8fw* pA = reinterpret_cast<const v8fw*>(hrow + base);
  // software pipeline: raw A floats for chunk c live across chunk c-1's WMMAs
  v8fw r0 = pA[0];      // K = base .. base+7
  v8fw r1 = pA[2];      // K = 16+base .. 16+base+7
  #pragma unroll
  for (int c = 0; c < 8; ++c) {
    union { v16bf v; v8bf h2[2]; } A;
    A.h2[0] = __builtin_convertvector(r0, v8bf);  // packed v_cvt_pk_bf16_f32
    A.h2[1] = __builtin_convertvector(r1, v8bf);
    if (c < 7) {
      r0 = pA[(c + 1) * 4];
      r1 = pA[(c + 1) * 4 + 2];
    }
    if (c < 6) __builtin_prefetch(hrow + (c + 2) * 32 + base, 0, 3); // WGP-scope stream-ahead
    #pragma unroll
    for (int t = 0; t < 4; ++t) {
      v16bf B = bp[(c * 4 + t) * 32 + lane];
      acc[t] = __builtin_amdgcn_wmma_f32_16x16x32_bf16(
          false, A.v, false, B, (short)0, acc[t], false, false);
    }
  }

  float asv[4], adv[4], bv[4];
  #pragma unroll
  for (int t = 0; t < 4; ++t) {
    asv[t] = a_src[t * 16 + m];
    adv[t] = a_dest[t * 16 + m];
    bv[t]  = b[t * 16 + m];
  }

  // C/D layout: lane group hi, VGPR v -> row = v + 8*hi; col = t*16 + m
  #pragma unroll
  for (int v = 0; v < 8; ++v) {
    const int r = row0 + v + 8 * hi;
    float s1 = 0.f, s2 = 0.f;
    #pragma unroll
    for (int t = 0; t < 4; ++t) {
      float val = acc[t][v] + bv[t];
      if (r < N) Wh[(long long)r * 64 + t * 16 + m] = val;
      s1 += val * asv[t];
      s2 += val * adv[t];
    }
    #pragma unroll
    for (int off = 1; off < 16; off <<= 1) {  // reduce 16 cols held by 16 lanes
      s1 += __shfl_xor(s1, off, 32);
      s2 += __shfl_xor(s2, off, 32);
    }
    if (m == 0 && r < N) { f1[r] = s1; f2[r] = s2; }
  }
}

// ---------------- Kernel 0: zero/init accumulators -------------------------
__global__ void gat_init(float* __restrict__ out, float* __restrict__ denom,
                         unsigned* __restrict__ segmax, int N, long long total)
{
  long long i = (long long)blockIdx.x * 256 + threadIdx.x;
  if (i < total) out[i] = 0.f;
  if (i < N) { denom[i] = 0.f; segmax[i] = 0u; } // ord 0 == "-inf"
}

// ---------------- Kernel 2: e = lrelu(f1[s]+f2[d]); segment max ------------
__global__ void gat_edge_max(const float* __restrict__ f1, const float* __restrict__ f2,
                             const int* __restrict__ es, const int* __restrict__ ed,
                             float* __restrict__ e, unsigned* __restrict__ segmax, int E)
{
  int i = blockIdx.x * 256 + threadIdx.x;
  if (i >= E) return;
  int s = es[i], d = ed[i];
  float x = f1[s] + f2[d];
  float v = x > 0.f ? x : LRELU_ALPHA * x;
  e[i] = v;
  atomicMax(segmax + s, f2ord(v));
}

// ---------------- Kernel 3: ex = exp(e - max[s]); segment sum --------------
__global__ void gat_edge_expsum(const int* __restrict__ es, float* __restrict__ e,
                                const unsigned* __restrict__ segmax,
                                float* __restrict__ denom, int E)
{
  int i = blockIdx.x * 256 + threadIdx.x;
  if (i >= E) return;
  int s = es[i];
  float ex = expf(e[i] - ord2f(segmax[s]));
  e[i] = ex;
  atomicAdd(denom + s, ex);
}

// ---------------- Kernel 4: SpMM scatter: out[s] += attn * Wh[d] -----------
// 32 lanes per edge; lane handles cols 2*lane, 2*lane+1 (float2 gather, L2-resident Wh)
__global__ void gat_edge_scatter(const int* __restrict__ es, const int* __restrict__ ed,
                                 const float* __restrict__ e, const float* __restrict__ denom,
                                 const float* __restrict__ Wh, float* __restrict__ out, int E)
{
  long long gid = (long long)blockIdx.x * 256 + threadIdx.x;
  int edge = (int)(gid >> 5);
  int lane = (int)(gid & 31);
  if (edge >= E) return;
  int s = es[edge], d = ed[edge];
  float attn = e[edge] / denom[s];
  float2 w = reinterpret_cast<const float2*>(Wh + (long long)d * 64)[lane];
  float* o = out + (long long)s * 64 + 2 * lane;
  atomicAdd(o,     attn * w.x);
  atomicAdd(o + 1, attn * w.y);
}

// ---------------- Kernel 5: ELU in place -----------------------------------
__global__ void gat_elu(float* __restrict__ out, long long total)
{
  long long i = (long long)blockIdx.x * 256 + threadIdx.x;
  if (i >= total) return;
  float x = out[i];
  out[i] = x > 0.f ? x : (expf(x) - 1.f);
}

extern "C" void kernel_launch(void* const* d_in, const int* in_sizes, int n_in,
                              void* d_out, int out_size, void* d_ws, size_t ws_size,
                              hipStream_t stream)
{
  const float* h      = (const float*)d_in[0];
  const float* W      = (const float*)d_in[1];
  const float* b      = (const float*)d_in[2];
  const float* a_src  = (const float*)d_in[3];
  const float* a_dest = (const float*)d_in[4];
  const int*   es     = (const int*)d_in[5];
  const int*   ed     = (const int*)d_in[6];
  const int N = in_sizes[0] / 256; // F_IN = 256
  const int E = in_sizes[5];
  float* out = (float*)d_out;

  // workspace carve-out (~40 MB): Wh | f1 | f2 | segmax | denom | e
  char* ws = (char*)d_ws;
  float*    Wh     = (float*)ws;     ws += (size_t)N * 64 * sizeof(float);
  float*    f1     = (float*)ws;     ws += (size_t)N * sizeof(float);
  float*    f2     = (float*)ws;     ws += (size_t)N * sizeof(float);
  unsigned* segmax = (unsigned*)ws;  ws += (size_t)N * sizeof(unsigned);
  float*    denom  = (float*)ws;     ws += (size_t)N * sizeof(float);
  float*    ebuf   = (float*)ws;

  const long long total = (long long)N * 64;
  gat_init<<<(unsigned)((total + 255) / 256), 256, 0, stream>>>(out, denom, segmax, N, total);
  gat_gemm_wmma<<<(N + 127) / 128, 256, 0, stream>>>(h, W, b, a_src, a_dest, Wh, f1, f2, N);
  gat_edge_max<<<(E + 255) / 256, 256, 0, stream>>>(f1, f2, es, ed, ebuf, segmax, E);
  gat_edge_expsum<<<(E + 255) / 256, 256, 0, stream>>>(es, ebuf, segmax, denom, E);
  const long long sthreads = (long long)E * 32;
  gat_edge_scatter<<<(unsigned)((sthreads + 255) / 256), 256, 0, stream>>>(es, ed, ebuf, denom, Wh, out, E);
  gat_elu<<<(unsigned)((total + 255) / 256), 256, 0, stream>>>(out, total);
}
